// RCSCNet_1460288880846
// MI455X (gfx1250) — matrix operand order repaired and checked by reading, not compiled, not run on hardware
//
#include <hip/hip_runtime.h>
#include <hip/hip_bf16.h>

// ---------------------------------------------------------------------------
// Problem constants (from reference): KS=7, KC=64, S=31, B=1, H=W=128, ITERS=3
// ---------------------------------------------------------------------------
#define KS    7
#define TAPS  49
#define KC    64
#define NS    31
#define HH    128
#define WW    128
#define NPIX  (HH * WW)                 // 16384
#define BAND  (KC * NPIX)               // 1048576 floats per band (per tensor)
#define TOTE  (NS * BAND)               // 32505856 elems of csc/tmp/zpre
#define XT    (WW + KS - 1)             // 134: full row + halo
#define LAMV  0.1f
#define EPSV  1e-5f

typedef __attribute__((ext_vector_type(16))) __bf16 v16bf;
typedef __attribute__((ext_vector_type(8)))  float  v8f;
typedef unsigned int u32x4 __attribute__((ext_vector_type(4)));
typedef int          i32x8 __attribute__((ext_vector_type(8)));
typedef int          i32x4 __attribute__((ext_vector_type(4)));

__device__ __forceinline__ unsigned short f32_to_bf16(float f) {
    unsigned int u = __float_as_uint(f);
    u += 0x7FFFu + ((u >> 16) & 1u);        // round-to-nearest-even
    return (unsigned short)(u >> 16);
}

__device__ __forceinline__ float softshrink(float x) {
    return (x > LAMV) ? (x - LAMV) : ((x < -LAMV) ? (x + LAMV) : 0.0f);
}

// ---------------------------------------------------------------------------
// Pack W_z (3 layers, [oc=64][ic=64][tap=49] f32) into bf16 A-operand tiles in
// a contiguous-k 16x32 A-operand lane order (the B reads use the same k order).
// Packed layout: [lyr][mtile(4)][tap(49)][kc2(2)][lane(32)][elem(16)] bf16
// Lane 0-15 : M=lane,    elem e -> K = e
// Lane16-31 : M=lane-16, elem e -> K = e+16
// ---------------------------------------------------------------------------
#define PACK_PER_LAYER (4 * TAPS * 2 * 32 * 16)   // 200704 bf16 elems

__global__ __launch_bounds__(256) void pack_wz_kernel(
        const float* __restrict__ Wz, unsigned short* __restrict__ packA) {
    int tid = blockIdx.x * 256 + threadIdx.x;
    if (tid >= 3 * PACK_PER_LAYER) return;
    int e    = tid & 15;  int r = tid >> 4;
    int lane = r & 31;    r >>= 5;
    int kc2  = r & 1;     r >>= 1;
    int t    = r % TAPS;  r /= TAPS;
    int mt   = r & 3;     r >>= 2;
    int lyr  = r;
    int k  = (lane < 16) ? e : e + 16;
    int oc = mt * 16 + (lane & 15);
    int ic = kc2 * 32 + k;
    packA[tid] = f32_to_bf16(Wz[(((size_t)lyr * KC + oc) * KC + ic) * TAPS + t]);
}

// ---------------------------------------------------------------------------
// 1->64 channel 7x7 SAME conv + optional add + softshrink.
// src: [s][y][x]; Wt: [64][49]; out: [s][c][y][x]  (f32)
// out16 (optional): channel-last bf16 copy [s][y][x][c] for the TDM/WMMA path.
// ---------------------------------------------------------------------------
__global__ __launch_bounds__(256) void conv1to64_kernel(
        const float* __restrict__ src, const float* __restrict__ Wt,
        const float* __restrict__ bias, const float* __restrict__ addend,
        float* __restrict__ out, unsigned short* __restrict__ out16) {
    int idx = blockIdx.x * 256 + threadIdx.x;            // < NS*KC*NPIX
    int x = idx & 127, y = (idx >> 7) & 127, c = (idx >> 14) & 63, s = idx >> 20;
    const float* sp = src + (size_t)s * NPIX;
    const float* wp = Wt + c * TAPS;
    float acc = bias[c];
    #pragma unroll
    for (int t = 0; t < TAPS; ++t) {
        int gy = y + t / KS - 3, gx = x + t % KS - 3;
        if (gy >= 0 && gy < HH && gx >= 0 && gx < WW)
            acc += wp[t] * sp[gy * WW + gx];
    }
    if (addend) acc += addend[idx];
    float v = softshrink(acc);
    out[idx] = v;
    if (out16) {
        int p = idx & (NPIX - 1);                        // y*128+x
        out16[((size_t)s * NPIX + p) * KC + c] = f32_to_bf16(v);
    }
}

// ---------------------------------------------------------------------------
// 64->1 channel 7x7 SAME conv.  mode=1: out = xin - (conv+b)   (residual)
//                               mode=0: out = conv + b         (final decode)
// ---------------------------------------------------------------------------
__global__ __launch_bounds__(256) void conv64to1_kernel(
        const float* __restrict__ src, const float* __restrict__ Wt,
        const float* __restrict__ bias1, const float* __restrict__ xin,
        float* __restrict__ out, int mode) {
    int idx = blockIdx.x * 256 + threadIdx.x;            // < NS*NPIX
    int x = idx & 127, y = (idx >> 7) & 127, s = idx >> 14;
    const float* sp = src + (size_t)s * BAND;
    float acc = bias1[0];
    for (int c = 0; c < KC; ++c) {
        const float* cp = sp + c * NPIX;
        const float* wp = Wt + c * TAPS;
        #pragma unroll
        for (int t = 0; t < TAPS; ++t) {
            int gy = y + t / KS - 3, gx = x + t % KS - 3;
            if (gy >= 0 && gy < HH && gx >= 0 && gx < WW)
                acc += wp[t] * cp[gy * WW + gx];
        }
    }
    out[idx] = mode ? (xin[idx] - acc) : acc;
}

// ---------------------------------------------------------------------------
// Heavy 64->64 7x7 conv as tap-decomposed implicit GEMM on
// v_wmma_f32_16x16x32_bf16.  One workgroup = one (band, row): 64 out-chan x
// 128 pixels.  Activations arrive as channel-last bf16 [s][y][x][c]; each of
// the 7 halo rows is one contiguous 16 KB run, fetched by the Tensor Data
// Mover directly into LDS (tensor_load_to_lds + s_wait_tensorcnt).  LDS tile:
// sh[(dy*134+dxx)*64 + c], 117 KB dynamic LDS.  8 waves: mtile = wid&3 (16 out
// chans), half = wid>>2 (64 px); 4 accumulators per wave so every A-operand
// (weight tile) is reused by 4 WMMAs.
// ---------------------------------------------------------------------------
__global__ __launch_bounds__(256) void zconv_wmma_kernel(
        const unsigned short* __restrict__ tmp16,
        const unsigned short* __restrict__ packA,
        const float* __restrict__ bz, float* __restrict__ zpre) {
    extern __shared__ __align__(32) unsigned short sh[];   // KS*XT*KC bf16

    const int y = blockIdx.x;
    const int s = blockIdx.y;

    // Zero only the regions the TDM will NOT write: the 3-column halos on each
    // side of every row, and the full interior of out-of-range rows.
    for (int j = threadIdx.x; j < KS * 6 * KC; j += 256) {
        int c = j & 63; int r2 = j >> 6;
        int col = r2 % 6; int dy = r2 / 6;
        int dxx = (col < 3) ? col : (WW + 3 + col - 3);    // {0,1,2,131,132,133}
        sh[(dy * XT + dxx) * KC + c] = 0;
    }
    #pragma unroll
    for (int dy = 0; dy < KS; ++dy) {
        int gy = y + dy - 3;
        if (gy < 0 || gy >= HH) {
            for (int j = threadIdx.x; j < WW * KC; j += 256)
                sh[(dy * XT + 3) * KC + j] = 0;
        }
    }
    __syncthreads();

    // Wave 0: issue one TDM descriptor per in-range halo row, then drain
    // TENSORcnt before releasing the block.
    if (threadIdx.x < 32) {
        #pragma unroll
        for (int dy = 0; dy < KS; ++dy) {
            int gy = y + dy - 3;
            if (gy < 0 || gy >= HH) continue;
            unsigned long long ga = (unsigned long long)(const void*)
                (tmp16 + ((size_t)s * NPIX + (size_t)gy * WW) * KC);
            unsigned int la = (unsigned int)(((dy * XT + 3) * KC) * 2);

            // D# group 0: count=1 | lds_addr | global_addr | type=2
            u32x4 g0;
            g0[0] = 1u;                                    // count=1, user mode
            g0[1] = la;                                    // lds_addr (bytes)
            g0[2] = (unsigned int)ga;                      // global_addr[31:0]
            g0[3] = (unsigned int)(ga >> 32) | (2u << 30); // addr[56:32] | type=2

            // D# group 1: data_size=2B; tensor_dim0 = tile_dim0 = 8192 elems
            // (one contiguous 128px x 64ch row); tensor_dim1 = tile_dim1 = 1.
            i32x8 g1;
            g1[0] = 0x00010000;              // workgroup_mask=0, data_size=1 (2B)
            g1[1] = (int)(8192u << 16);      // tensor_dim0 low16 -> bits 63:48
            g1[2] = 0x00010000;              // tensor_dim0 hi16=0 | tensor_dim1 lo16=1
            g1[3] = (int)(8192u << 16);      // tensor_dim1 hi=0 | tile_dim0=8192
            g1[4] = 1;                       // tile_dim1=1, tile_dim2=0
            g1[5] = 8192;                    // tensor_dim0_stride low
            g1[6] = 0;                       // stride0 hi | stride1 lo
            g1[7] = 0;                       // stride1 hi

            i32x4 g2; g2[0] = 1; g2[1] = 0; g2[2] = 0; g2[3] = 0; // dim2=1, tile_dim3=0
            i32x4 g3; g3[0] = 0; g3[1] = 0; g3[2] = 0; g3[3] = 0;
            i32x8 g4; g4[0] = 0; g4[1] = 0; g4[2] = 0; g4[3] = 0;
                      g4[4] = 0; g4[5] = 0; g4[6] = 0; g4[7] = 0;

            __builtin_amdgcn_tensor_load_to_lds(g0, g1, g2, g3, g4, 0);
        }
        __builtin_amdgcn_s_wait_tensorcnt(0);
    }
    __syncthreads();

    const int wid   = threadIdx.x >> 5;
    const int lane  = threadIdx.x & 31;
    const int mtile = wid & 3;
    const int half  = wid >> 2;              // pixel block: half*64 .. +63
    const int ln16  = lane & 15;
    const int chi   = (lane & 16) ? 16 : 0;

    v8f acc0 = {}, acc1 = {}, acc2 = {}, acc3 = {};
    const unsigned short* Ab = packA + (size_t)mtile * TAPS * 2 * 32 * 16;

    for (int t = 0; t < TAPS; ++t) {
        const int dy = t / KS, dx = t % KS;
        const int rowoff = (dy * XT + dx + half * 64 + ln16) * KC;
        #pragma unroll
        for (int kc2 = 0; kc2 < 2; ++kc2) {
            const int cbase = kc2 * 32 + chi;
            v16bf am = *(const v16bf*)(const void*)&Ab[((t * 2 + kc2) * 32 + lane) * 16];
            v16bf b0 = *(const v16bf*)(const void*)&sh[rowoff + cbase];
            v16bf b1 = *(const v16bf*)(const void*)&sh[rowoff + 16 * KC + cbase];
            v16bf b2 = *(const v16bf*)(const void*)&sh[rowoff + 32 * KC + cbase];
            v16bf b3 = *(const v16bf*)(const void*)&sh[rowoff + 48 * KC + cbase];
            acc0 = __builtin_amdgcn_wmma_f32_16x16x32_bf16(
                       false, am, false, b0, (short)0, acc0, false, false);
            acc1 = __builtin_amdgcn_wmma_f32_16x16x32_bf16(
                       false, am, false, b1, (short)0, acc1, false, false);
            acc2 = __builtin_amdgcn_wmma_f32_16x16x32_bf16(
                       false, am, false, b2, (short)0, acc2, false, false);
            acc3 = __builtin_amdgcn_wmma_f32_16x16x32_bf16(
                       false, am, false, b3, (short)0, acc3, false, false);
        }
    }

    // C/D layout: VGPR r -> M=r (lanes 0-15) / M=r+8 (lanes 16-31), N=lane%16
    const int xb = half * 64 + ln16;
    #pragma unroll
    for (int r = 0; r < 8; ++r) {
        int m  = (lane < 16) ? r : (r + 8);
        int oc = mtile * 16 + m;
        float bb = bz[oc];
        size_t base = ((size_t)s * KC + oc) * NPIX + y * WW + xb;
        zpre[base +  0] = acc0[r] + bb;
        zpre[base + 16] = acc1[r] + bb;
        zpre[base + 32] = acc2[r] + bb;
        zpre[base + 48] = acc3[r] + bb;
    }
}

// ---------------------------------------------------------------------------
// BatchNorm statistics per (band, channel) over 16384 pixels.
// stats[2*sc] = mean, stats[2*sc+1] = rsqrt(var+eps)
// ---------------------------------------------------------------------------
__global__ __launch_bounds__(256) void bn_stats_kernel(
        const float* __restrict__ zpre, float* __restrict__ stats) {
    int sc = blockIdx.x;                                  // s*64+c
    const float* p = zpre + (size_t)sc * NPIX;
    float sum = 0.f, sq = 0.f;
    for (int i = threadIdx.x; i < NPIX; i += 256) { float v = p[i]; sum += v; sq += v * v; }
    __shared__ float ssum[256], ssq[256];
    ssum[threadIdx.x] = sum; ssq[threadIdx.x] = sq;
    __syncthreads();
    for (int off = 128; off > 0; off >>= 1) {
        if (threadIdx.x < off) {
            ssum[threadIdx.x] += ssum[threadIdx.x + off];
            ssq[threadIdx.x]  += ssq[threadIdx.x + off];
        }
        __syncthreads();
    }
    if (threadIdx.x == 0) {
        float mu  = ssum[0] * (1.0f / NPIX);
        float var = ssq[0] * (1.0f / NPIX) - mu * mu;
        stats[2 * sc]     = mu;
        stats[2 * sc + 1] = rsqrtf(var + EPSV);
    }
}

// z = sigmoid(BN(zpre)*gamma + beta), in place.
__global__ __launch_bounds__(256) void gate_kernel(
        float* __restrict__ zpre, const float* __restrict__ stats,
        const float* __restrict__ gamma, const float* __restrict__ beta) {
    int idx = blockIdx.x * 256 + threadIdx.x;            // < TOTE
    int sc = idx >> 14;  int c = sc & 63;
    float zn = (zpre[idx] - stats[2 * sc]) * stats[2 * sc + 1];
    float a  = zn * gamma[c] + beta[c];
    zpre[idx] = 1.0f / (1.0f + __expf(-a));
}

// Directional GRU-style scan over the 31 bands; 1 thread per (channel,pixel).
__global__ __launch_bounds__(256) void scan_kernel(
        const float* __restrict__ tmp, const float* __restrict__ z,
        float* __restrict__ csc, int fwd) {
    int q = blockIdx.x * 256 + threadIdx.x;              // < BAND
    if (fwd) {
        float carry = tmp[q];
        csc[q] = carry;
        for (int s = 1; s < NS; ++s) {
            size_t o = (size_t)s * BAND + q;
            float zi = z[o];
            carry = zi * carry + (1.0f - zi) * tmp[o];
            csc[o] = carry;
        }
    } else {
        size_t last = (size_t)(NS - 1) * BAND + q;
        float carry = tmp[last];
        csc[last] = carry;
        for (int s = NS - 2; s >= 0; --s) {
            size_t o = (size_t)s * BAND + q;
            float zi = z[o];
            carry = zi * carry + (1.0f - zi) * tmp[o];
            csc[o] = carry;
        }
    }
}

// ---------------------------------------------------------------------------
extern "C" void kernel_launch(void* const* d_in, const int* in_sizes, int n_in,
                              void* d_out, int out_size, void* d_ws, size_t ws_size,
                              hipStream_t stream) {
    const float* inputs = (const float*)d_in[0];   // [1,31,128,128]
    const float* W_fe   = (const float*)d_in[1];   // [64,1,7,7]
    const float* b_fe   = (const float*)d_in[2];   // [64]
    const float* W_enc  = (const float*)d_in[3];   // [3,64,1,7,7]
    const float* b_enc  = (const float*)d_in[4];   // [3,64]
    const float* W_dec  = (const float*)d_in[5];   // [3,1,64,7,7]
    const float* b_dec  = (const float*)d_in[6];   // [3,1]
    const float* W_last = (const float*)d_in[7];   // [1,64,7,7]
    const float* b_last = (const float*)d_in[8];   // [1]
    const float* W_z    = (const float*)d_in[9];   // [3,64,64,7,7]
    const float* b_z    = (const float*)d_in[10];  // [3,64]
    const float* bn_g   = (const float*)d_in[11];  // [3,64]
    const float* bn_b   = (const float*)d_in[12];  // [3,64]

    // Workspace (floats): csc | tmp | zpre | res | stats | packA(bf16) | tmp16(bf16)
    float* ws    = (float*)d_ws;
    float* csc   = ws;                       // TOTE
    float* tmp   = ws + (size_t)TOTE;        // TOTE
    float* zpre  = ws + (size_t)2 * TOTE;    // TOTE
    float* res   = ws + (size_t)3 * TOTE;    // NS*NPIX
    float* stats = res + (size_t)NS * NPIX;  // NS*KC*2
    unsigned short* packA =
        (unsigned short*)(stats + (size_t)NS * KC * 2);      // 3*PACK_PER_LAYER
    unsigned short* tmp16 = packA + (size_t)3 * PACK_PER_LAYER; // TOTE bf16

    const int gridBig  = TOTE / 256;         // 126976
    const int gridPix  = NS * NPIX / 256;    // 1984
    const int gridScan = BAND / 256;         // 4096
    const size_t zconv_lds = (size_t)KS * XT * KC * sizeof(unsigned short); // 120064 B

    // Pack bf16 weights for all 3 layers.
    pack_wz_kernel<<<(3 * PACK_PER_LAYER + 255) / 256, 256, 0, stream>>>(W_z, packA);

    // Initial sparse code: softshrink(conv_fe(x)).
    conv1to64_kernel<<<gridBig, 256, 0, stream>>>(inputs, W_fe, b_fe, nullptr, csc, nullptr);

    for (int it = 0; it < 3; ++it) {
        // res = x - dec(csc)
        conv64to1_kernel<<<gridPix, 256, 0, stream>>>(
            csc, W_dec + (size_t)it * KC * TAPS, b_dec + it, inputs, res, 1);
        // tmp = softshrink(csc + enc(res)); also emit channel-last bf16 copy
        conv1to64_kernel<<<gridBig, 256, 0, stream>>>(
            res, W_enc + (size_t)it * KC * TAPS, b_enc + (size_t)it * KC, csc, tmp, tmp16);
        // zpre = conv_z(tmp)  -- TDM staging + WMMA, one workgroup per (row, band)
        zconv_wmma_kernel<<<dim3(HH, NS), 256, zconv_lds, stream>>>(
            tmp16, packA + (size_t)it * PACK_PER_LAYER, b_z + (size_t)it * KC, zpre);
        // BN stats + sigmoid gate (in place on zpre)
        bn_stats_kernel<<<NS * KC, 256, 0, stream>>>(zpre, stats);
        gate_kernel<<<gridBig, 256, 0, stream>>>(
            zpre, stats, bn_g + (size_t)it * KC, bn_b + (size_t)it * KC);
        // directional band scan -> csc
        scan_kernel<<<gridScan, 256, 0, stream>>>(tmp, zpre, csc, (it & 1) == 0);
    }

    // Outputs: tuple (csc, outputs) flattened into d_out.
    (void)hipMemcpyAsync(d_out, csc, (size_t)TOTE * sizeof(float),
                         hipMemcpyDeviceToDevice, stream);
    conv64to1_kernel<<<gridPix, 256, 0, stream>>>(
        csc, W_last, b_last, nullptr, (float*)d_out + (size_t)TOTE, 0);
}
